// LlamaAttention_14740327760023
// MI455X (gfx1250) — matrix operand order repaired and checked by reading, not compiled, not run on hardware
//
#include <hip/hip_runtime.h>

// ---------------------------------------------------------------------------
// Types for CDNA5 WMMA fragments
// ---------------------------------------------------------------------------
typedef __attribute__((ext_vector_type(2)))  float  v2f;
typedef __attribute__((ext_vector_type(4)))  float  v4f;
typedef __attribute__((ext_vector_type(8)))  float  v8f;
typedef __attribute__((ext_vector_type(8)))  __bf16 v8bf;
typedef __attribute__((ext_vector_type(16))) __bf16 v16bf;

// gcc-style vector type matching the async-LDS builtin's parameter type
typedef int i32x4 __attribute__((__vector_size__(16)));

#define NUM_HEADS 32
#define HEAD_DIM  128
#define SEQ_LEN   2048
#define BATCH     2
#define HID       4096
#define QKV_W     (3 * HID)         // 12288
#define ROWS      (BATCH * SEQ_LEN) // 4096

// ---------------------------------------------------------------------------
// CDNA5 async global->LDS copy (GLOBAL_LOAD_ASYNC_TO_LDS_B128, ASYNCcnt).
// Guarded: falls back to a plain vector copy if the builtin is absent.
// ---------------------------------------------------------------------------
#if defined(__has_builtin)
#  if __has_builtin(__builtin_amdgcn_global_load_async_to_lds_b128)
#    define HAVE_ASYNC_LDS 1
#  endif
#endif

__device__ __forceinline__ void async_cp_b128(const float* g, float* l) {
#if defined(HAVE_ASYNC_LDS)
    __builtin_amdgcn_global_load_async_to_lds_b128(
        (__attribute__((address_space(1))) i32x4*)g,
        (__attribute__((address_space(3))) i32x4*)l,
        0, 0);
#else
    *(v4f*)l = *(const v4f*)g;
#endif
}

__device__ __forceinline__ void async_wait0() {
#if defined(HAVE_ASYNC_LDS)
#  if __has_builtin(__builtin_amdgcn_s_wait_asynccnt)
    __builtin_amdgcn_s_wait_asynccnt(0);
#  else
    asm volatile("s_wait_asynccnt 0x0" ::: "memory");
#  endif
#endif
}

// ---------------------------------------------------------------------------
// Kernel 1/4: exact-fp32 GEMM via V_WMMA_F32_16X16X4_F32, LDS-tiled.
// C[M,N] = A[M,K] @ B[K,N], row-major fp32. Requires M%64==0, N%128==0, K%32==0.
// Block = 256 threads = 8 waves (2x4). Block tile 64x128, wave tile 32x32
// (2x2 WMMA fragments). K-chunks of 32 double-buffered in LDS via async copy.
//   A-frag (16x4 f32): lanes 0-15 hold {K=0,1}, lanes 16-31 hold {K=2,3}.
//   B-frag (4x16 f32): VGPR0/1 hold K={0,1} (lanes 0-15), K={2,3} (lanes 16-31).
// ---------------------------------------------------------------------------
#define KC      32
#define A_STR   36   // 64 rows x (32+4) floats: 16B-aligned rows, conflict-free
#define B_STR   132  // 32 rows x (128+4) floats: 16B-aligned rows

__global__ __launch_bounds__(256) void gemm_f32_wmma(const float* __restrict__ A,
                                                     const float* __restrict__ B,
                                                     float* __restrict__ C,
                                                     int M, int N, int K) {
    __shared__ __align__(16) float As[2][64 * A_STR];
    __shared__ __align__(16) float Bs[2][KC * B_STR];

    const int tid   = threadIdx.x;
    const int wave  = tid >> 5;
    const int lane  = tid & 31;
    const int lhalf = lane >> 4;
    const int lmod  = lane & 15;
    const int wr    = (wave >> 2) * 32;          // wave row offset in block tile
    const int wc    = (wave & 3) * 32;           // wave col offset in block tile
    const int brow  = blockIdx.y * 64;
    const int bcol  = blockIdx.x * 128;

    // ---- per-thread copy descriptors ----
    // A tile: 64 rows x 32 floats = 512 b128 segments, 2 per thread.
    const int as0 = tid * 2, as1 = tid * 2 + 1;
    const int ar0 = as0 >> 3, ac0 = (as0 & 7) * 4;
    const int ar1 = as1 >> 3, ac1 = (as1 & 7) * 4;
    // B tile: 32 rows x 128 floats = 1024 b128 segments, 4 per thread.
    int brw[4], bcl[4];
#pragma unroll
    for (int q = 0; q < 4; ++q) {
        const int s = tid * 4 + q;
        brw[q] = s >> 5;
        bcl[q] = (s & 31) * 4;
    }

    auto load_chunk = [&](int k0, int buf) {
        async_cp_b128(A + (size_t)(brow + ar0) * K + k0 + ac0,
                      &As[buf][ar0 * A_STR + ac0]);
        async_cp_b128(A + (size_t)(brow + ar1) * K + k0 + ac1,
                      &As[buf][ar1 * A_STR + ac1]);
#pragma unroll
        for (int q = 0; q < 4; ++q)
            async_cp_b128(B + (size_t)(k0 + brw[q]) * N + bcol + bcl[q],
                          &Bs[buf][brw[q] * B_STR + bcl[q]]);
    };

    v8f acc[2][2] = {};

    load_chunk(0, 0);
    async_wait0();
    __syncthreads();

    const int nk = K / KC;
    for (int t = 0; t < nk; ++t) {
        const int cur = t & 1;
        if (t + 1 < nk) load_chunk((t + 1) * KC, cur ^ 1);   // overlap with compute

        const float* as = &As[cur][0];
        const float* bs = &Bs[cur][0];
#pragma unroll
        for (int kk = 0; kk < KC; kk += 4) {
            v2f a0 = *(const v2f*)(as + (wr + lmod) * A_STR + kk + lhalf * 2);
            v2f a1 = *(const v2f*)(as + (wr + 16 + lmod) * A_STR + kk + lhalf * 2);
            v2f b0, b1;
            b0.x = bs[(kk + lhalf * 2)     * B_STR + wc + lmod];
            b0.y = bs[(kk + lhalf * 2 + 1) * B_STR + wc + lmod];
            b1.x = bs[(kk + lhalf * 2)     * B_STR + wc + 16 + lmod];
            b1.y = bs[(kk + lhalf * 2 + 1) * B_STR + wc + 16 + lmod];
            acc[0][0] = __builtin_amdgcn_wmma_f32_16x16x4_f32(
                false, a0, false, b0, (short)0, acc[0][0], false, false);
            acc[0][1] = __builtin_amdgcn_wmma_f32_16x16x4_f32(
                false, a0, false, b1, (short)0, acc[0][1], false, false);
            acc[1][0] = __builtin_amdgcn_wmma_f32_16x16x4_f32(
                false, a1, false, b0, (short)0, acc[1][0], false, false);
            acc[1][1] = __builtin_amdgcn_wmma_f32_16x16x4_f32(
                false, a1, false, b1, (short)0, acc[1][1], false, false);
        }
        async_wait0();
        __syncthreads();
    }

    // ---- store 4 fragments ----
#pragma unroll
    for (int i = 0; i < 2; ++i) {
#pragma unroll
        for (int j = 0; j < 2; ++j) {
            float* crow = C + (size_t)(brow + wr + 16 * i + lhalf * 8) * N
                            + bcol + wc + 16 * j + lmod;
#pragma unroll
            for (int r = 0; r < 8; ++r)
                crow[(size_t)r * N] = acc[i][j][r];
        }
    }
}

// ---------------------------------------------------------------------------
// Kernel 2/4: RoPE applied in-place to q and k inside the qkv buffer.
// ---------------------------------------------------------------------------
__global__ __launch_bounds__(256) void rope_kernel(float* __restrict__ qkv) {
    size_t idx = (size_t)blockIdx.x * blockDim.x + threadIdx.x;
    const size_t total = (size_t)BATCH * SEQ_LEN * NUM_HEADS * 64;
    if (idx >= total) return;

    const int i = idx & 63;
    size_t t   = idx >> 6;
    const int h = t & 31;
    t >>= 5;
    const int s = t & (SEQ_LEN - 1);
    const int b = (int)(t >> 11);

    const float inv_freq = __powf(10000.0f, -(float)i * (1.0f / 64.0f));
    const float ang = (float)s * inv_freq;
    const float c = cosf(ang), sn = sinf(ang);

    float* base = qkv + ((size_t)(b * SEQ_LEN + s)) * QKV_W + (size_t)h * HEAD_DIM;
    float q1 = base[i], q2 = base[i + 64];
    base[i]      = q1 * c - q2 * sn;
    base[i + 64] = q2 * c + q1 * sn;
    float* kb = base + HID;
    float k1 = kb[i], k2 = kb[i + 64];
    kb[i]      = k1 * c - k2 * sn;
    kb[i + 64] = k2 * c + k1 * sn;
}

// ---------------------------------------------------------------------------
// Kernel 3/4: causal flash attention (bf16 WMMA, fp32 accumulate).
// Grid: (S/64, NUM_HEADS, BATCH). Block: 128 threads = 4 waves.
// ---------------------------------------------------------------------------
__global__ __launch_bounds__(128) void flash_attn(const float* __restrict__ qkv,
                                                  float* __restrict__ ctx) {
    const int qblk = blockIdx.x, h = blockIdx.y, b = blockIdx.z;
    const int wave  = threadIdx.x >> 5;
    const int lane  = threadIdx.x & 31;
    const int lhalf = lane >> 4;
    const int lmod  = lane & 15;
    const int qbase = qblk * 64 + wave * 16;

    __shared__ __align__(64) __bf16 Kt[32][HEAD_DIM];     // [key][d]
    __shared__ __align__(64) __bf16 Vt[HEAD_DIM][32];     // [d][key]  (transposed)
    __shared__ __align__(64) float  Sc[4][16][33];        // per-wave score staging
    __shared__ __align__(64) __bf16 Pst[4][16][32];       // per-wave P staging
    __shared__ __align__(64) float  rstat[4][2][16];      // [wave][alpha|sum][row]

    const float scaling = 0.08838834764831845f;           // 1/sqrt(128)
    const float NEG = -1.0e30f;

    const float* qptr = qkv + ((size_t)(b * SEQ_LEN + qbase + lmod)) * QKV_W
                            + (size_t)h * HEAD_DIM;
    v16bf qf[4];
#pragma unroll
    for (int c = 0; c < 4; ++c) {
        const int d0 = c * 32 + lhalf * 16;
        v4f a0 = *(const v4f*)(qptr + d0);
        v4f a1 = *(const v4f*)(qptr + d0 + 4);
        v4f b0 = *(const v4f*)(qptr + d0 + 8);
        v4f b1 = *(const v4f*)(qptr + d0 + 12);
#pragma unroll
        for (int e = 0; e < 4; ++e) {
            qf[c][e]      = (__bf16)a0[e];
            qf[c][e + 4]  = (__bf16)a1[e];
            qf[c][e + 8]  = (__bf16)b0[e];
            qf[c][e + 12] = (__bf16)b1[e];
        }
    }

    v8f   ctxf[8] = {};
    float rmax = NEG, rsum = 0.0f;    // valid on lanes 0..15 (row = lane)

    const int ktiles = (qblk + 1) * 2;
    for (int kt = 0; kt < ktiles; ++kt) {
        const int kbase = kt * 32;
        {
            const int key = threadIdx.x >> 2;
            const int dq  = (threadIdx.x & 3) * 32;
            const float* kp = qkv + ((size_t)(b * SEQ_LEN + kbase + key)) * QKV_W
                                  + HID + (size_t)h * HEAD_DIM + dq;
            const float* vp = kp + HID;
#pragma unroll 8
            for (int d = 0; d < 32; ++d) {
                Kt[key][dq + d] = (__bf16)kp[d];
                Vt[dq + d][key] = (__bf16)vp[d];
            }
        }
        __syncthreads();

        if (kbase <= qbase + 15) {
            v8f s0 = {}, s1 = {};
#pragma unroll
            for (int c = 0; c < 4; ++c) {
                v16bf bk0 = *(const v16bf*)&Kt[lmod][c * 32 + lhalf * 16];
                v16bf bk1 = *(const v16bf*)&Kt[lmod + 16][c * 32 + lhalf * 16];
                s0 = __builtin_amdgcn_wmma_f32_16x16x32_bf16(
                    false, qf[c], false, bk0, (short)0, s0, false, false);
                s1 = __builtin_amdgcn_wmma_f32_16x16x32_bf16(
                    false, qf[c], false, bk1, (short)0, s1, false, false);
            }
#pragma unroll
            for (int i = 0; i < 8; ++i) {
                Sc[wave][i + lhalf * 8][lmod]      = s0[i];
                Sc[wave][i + lhalf * 8][lmod + 16] = s1[i];
            }

            if (lane < 16) {
                const int qrow = qbase + lane;
                float m = rmax;
#pragma unroll 8
                for (int j = 0; j < 32; ++j) {
                    float v = Sc[wave][lane][j] * scaling;
                    if (kbase + j > qrow) v = NEG;
                    m = fmaxf(m, v);
                }
                const float alpha = __expf(rmax - m);
                float sum = rsum * alpha;
#pragma unroll 8
                for (int j = 0; j < 32; ++j) {
                    float v = Sc[wave][lane][j] * scaling;
                    if (kbase + j > qrow) v = NEG;
                    float p = __expf(v - m);
                    Pst[wave][lane][j] = (__bf16)p;
                    sum += p;
                }
                rmax = m;
                rsum = sum;
                rstat[wave][0][lane] = alpha;
            }

#pragma unroll
            for (int i = 0; i < 8; ++i) {
                const float al = rstat[wave][0][lhalf * 8 + i];
#pragma unroll
                for (int j = 0; j < 8; ++j)
                    ctxf[j][i] *= al;
            }

            v8bf plo = *(const v8bf*)&Pst[wave][lmod][lhalf * 16];
            v8bf phi = *(const v8bf*)&Pst[wave][lmod][lhalf * 16 + 8];
            v16bf pf = __builtin_shufflevector(plo, phi,
                0, 1, 2, 3, 4, 5, 6, 7, 8, 9, 10, 11, 12, 13, 14, 15);

#pragma unroll
            for (int j = 0; j < 8; ++j) {
                v16bf vf = *(const v16bf*)&Vt[16 * j + lmod][lhalf * 16];
                ctxf[j] = __builtin_amdgcn_wmma_f32_16x16x32_bf16(
                    false, pf, false, vf, (short)0, ctxf[j], false, false);
            }
        }
        __syncthreads();
    }

    if (lane < 16) rstat[wave][1][lane] = rsum;
    float rs[8];
#pragma unroll
    for (int i = 0; i < 8; ++i)
        rs[i] = 1.0f / rstat[wave][1][lhalf * 8 + i];

    float* cp = ctx + ((size_t)(b * SEQ_LEN + qbase)) * HID + (size_t)h * HEAD_DIM;
#pragma unroll
    for (int j = 0; j < 8; ++j) {
#pragma unroll
        for (int i = 0; i < 8; ++i) {
            const int row = i + lhalf * 8;
            cp[(size_t)row * HID + j * 16 + lmod] = ctxf[j][i] * rs[i];
        }
    }
}

// ---------------------------------------------------------------------------
// Launch
// ---------------------------------------------------------------------------
extern "C" void kernel_launch(void* const* d_in, const int* in_sizes, int n_in,
                              void* d_out, int out_size, void* d_ws, size_t ws_size,
                              hipStream_t stream) {
    const float* x    = (const float*)d_in[0];   // [B,S,H]
    const float* Wqkv = (const float*)d_in[1];   // [H, 3H]
    const float* Wo   = (const float*)d_in[2];   // [H, H]
    float* out = (float*)d_out;                  // [B,S,H]

    float* qkv = (float*)d_ws;                                   // ROWS x 12288
    float* ctx = (float*)((char*)d_ws +
                          (size_t)ROWS * QKV_W * sizeof(float)); // ROWS x 4096

    // 1) qkv = x @ Wqkv   (exact fp32 WMMA, LDS double-buffered async pipeline)
    gemm_f32_wmma<<<dim3(QKV_W / 128, ROWS / 64), 256, 0, stream>>>(
        x, Wqkv, qkv, ROWS, QKV_W, HID);

    // 2) RoPE on q,k in place
    {
        const size_t total = (size_t)BATCH * SEQ_LEN * NUM_HEADS * 64;
        rope_kernel<<<(unsigned)((total + 255) / 256), 256, 0, stream>>>(qkv);
    }

    // 3) causal flash attention (bf16 WMMA, fp32 accumulate)
    flash_attn<<<dim3(SEQ_LEN / 64, NUM_HEADS, BATCH), 128, 0, stream>>>(qkv, ctx);

    // 4) out = ctx @ Wo   (exact fp32 WMMA)
    gemm_f32_wmma<<<dim3(HID / 128, ROWS / 64), 256, 0, stream>>>(
        ctx, Wo, out, ROWS, HID, HID);
}
